// FoldingBlock_45664092291509
// MI455X (gfx1250) — matrix operand order repaired and checked by reading, not compiled, not run on hardware
//
#include <hip/hip_runtime.h>
#include <hip/hip_bf16.h>

// ---------------------------------------------------------------------------
// MI455X (gfx1250) folding-block implementation.
// All GEMM work goes through v_wmma_f32_16x16x32_bf16 (wave32, one wave
// computes a 16x64 output strip: 4 accumulators, A fragment converted once
// per K-step, 4 back-to-back WMMAs). Softmax / LayerNorm in f32 VALU.
// Fragment layouts per CDNA5 ISA 7.12.2:
//   A (16x32 bf16): lane&15 = M row; lanes<16 hold K [k0..k0+7],[k0+16..k0+23],
//                   lanes>=16 hold K [k0+8..k0+15],[k0+24..k0+31]
//   B (32x16 bf16): lane&15 = N col; lanes<16 hold K [k0..k0+7],[k0+16..k0+23],
//                   lanes>=16 hold K [k0+8..k0+15],[k0+24..k0+31]   (B fed as W row-major NxK)
//   C/D (16x16 f32): lane&15 = N; VGPR r -> M = r + 8*(lane>=16)
// ---------------------------------------------------------------------------

typedef __attribute__((ext_vector_type(16))) __bf16 v16bf;
typedef __attribute__((ext_vector_type(8)))  float  v8f;

#define WMMA_BF16(ACC, A, B)                                                   \
  (ACC) = __builtin_amdgcn_wmma_f32_16x16x32_bf16(false, (A), false, (B),      \
                                                  (short)0, (ACC), false, false)

#define NSEQ 192
#define DSNG 384
#define DPR  128
#define NHEAD 8
#define CHEAD 32
#define HDIM 256                 // NHEAD*CHEAD
#define NPAIR (NSEQ * NSEQ)      // 36864
#define NEG_INF (-32768.0f)      // 2**15

__device__ __forceinline__ float sigf(float x) { return 1.0f / (1.0f + __expf(-x)); }

// ---- fragment loaders (f32 source, convert to bf16) -----------------------
__device__ __forceinline__ v16bf afrag_row(const float* __restrict__ row, int k0, int ah) {
  v16bf a;
#pragma unroll
  for (int e = 0; e < 8; ++e) {
    a[e]     = (__bf16)row[k0 + ah * 8 + e];
    a[8 + e] = (__bf16)row[k0 + 16 + ah * 8 + e];
  }
  return a;
}
__device__ __forceinline__ v16bf bfrag_row(const float* __restrict__ row, int k0, int ah) {
  v16bf b;
#pragma unroll
  for (int e = 0; e < 16; ++e) b[e] = (__bf16)row[k0 + (e >> 3) * 16 + ah * 8 + (e & 7)];
  return b;
}
__device__ __forceinline__ v16bf afrag_str(const float* __restrict__ base, int k0, int ah, int stride) {
  v16bf a;
#pragma unroll
  for (int e = 0; e < 8; ++e) {
    a[e]     = (__bf16)base[(size_t)(k0 + ah * 8 + e) * stride];
    a[8 + e] = (__bf16)base[(size_t)(k0 + 16 + ah * 8 + e) * stride];
  }
  return a;
}
__device__ __forceinline__ v16bf bfrag_str(const float* __restrict__ base, int k0, int ah, int stride) {
  v16bf b;
#pragma unroll
  for (int e = 0; e < 16; ++e) b[e] = (__bf16)base[(size_t)(k0 + (e >> 3) * 16 + ah * 8 + (e & 7)) * stride];
  return b;
}

// ---- LayerNorm over last dim (one row per block, 128 threads) --------------
__global__ void ln_kernel(const float* __restrict__ x, float* __restrict__ y, int D) {
  const int row = blockIdx.x;
  const float* xr = x + (size_t)row * D;
  float* yr = y + (size_t)row * D;
  float s = 0.f, ss = 0.f;
  for (int i = threadIdx.x; i < D; i += blockDim.x) { float v = xr[i]; s += v; ss += v * v; }
  __shared__ float shs[8], shss[8];
#pragma unroll
  for (int off = 16; off > 0; off >>= 1) { s += __shfl_down(s, off, 32); ss += __shfl_down(ss, off, 32); }
  const int wid = threadIdx.x >> 5;
  if ((threadIdx.x & 31) == 0) { shs[wid] = s; shss[wid] = ss; }
  __syncthreads();
  if (threadIdx.x == 0) {
    const int nw = blockDim.x >> 5;
    for (int w = 1; w < nw; ++w) { s += shs[w]; ss += shss[w]; }
    float mean = s / D;
    float var = ss / D - mean * mean;
    shs[0] = mean; shss[0] = rsqrtf(var + 1e-5f);
  }
  __syncthreads();
  const float mean = shs[0], rstd = shss[0];
  for (int i = threadIdx.x; i < D; i += blockDim.x) yr[i] = (xr[i] - mean) * rstd;
}

// ---- generic WMMA GEMM: Y = act(X(M,K) @ W(N,K)^T * scale + bias) (+res) ---
// one wave per 16x64 strip (4 WMMAs per K-step); grid = (N/64, M/16); K%32==0
__global__ void wgemm_kernel(const float* __restrict__ X, int ldx,
                             const float* __restrict__ W, int ldw,
                             const float* __restrict__ bias,
                             float* __restrict__ Y, int ldy,
                             int K, int act, float scale,
                             const float* __restrict__ res) {
  const int lane = threadIdx.x;
  const int m0 = blockIdx.y << 4, n0 = blockIdx.x << 6;
  const int am = lane & 15, ah = lane >> 4;
  const float* xrow = X + (size_t)(m0 + am) * ldx;
  const float* wrow0 = W + (size_t)(n0 + am) * ldw;
  v8f z = {};
  v8f acc[4] = {z, z, z, z};
  for (int k0 = 0; k0 < K; k0 += 32) {
    v16bf a = afrag_row(xrow, k0, ah);
#pragma unroll
    for (int t = 0; t < 4; ++t) {
      v16bf b = bfrag_row(wrow0 + (size_t)(t * 16) * ldw, k0, ah);
      WMMA_BF16(acc[t], a, b);
    }
  }
#pragma unroll
  for (int t = 0; t < 4; ++t) {
    const int cn = n0 + t * 16 + am;
    const float bv = bias ? bias[cn] : 0.0f;
#pragma unroll
    for (int r = 0; r < 8; ++r) {
      const int m = m0 + r + (ah << 3);
      float v = acc[t][r] * scale + bv;
      if (act == 1) v = fmaxf(v, 0.0f);
      else if (act == 2) v = sigf(v);
      if (res) v += res[(size_t)m * ldy + cn];
      Y[(size_t)m * ldy + cn] = v;
    }
  }
}

// ---- pair -> attention bias projection (N=8, f32-exact scalar dot) ---------
__global__ void bias_proj_kernel(const float* __restrict__ pln, const float* __restrict__ wb,
                                 const float* __restrict__ bb, float* __restrict__ biasb) {
  const int idx = blockIdx.x * blockDim.x + threadIdx.x;   // NPAIR*NHEAD threads
  if (idx >= NPAIR * NHEAD) return;
  const int row = idx >> 3, h = idx & 7;
  const float* x = pln + (size_t)row * DPR;
  const float* w = wb + (size_t)h * DPR;
  float s = bb[h];
  for (int kk = 0; kk < DPR; ++kk) s += x[kk] * w[kk];
  biasb[idx] = s;
}

// ---- mask2d[i,j] = mask[i]*mask[j] ----------------------------------------
__global__ void mask2d_kernel(const float* __restrict__ mask, float* __restrict__ m2) {
  m2[(size_t)blockIdx.x * NSEQ + threadIdx.x] = mask[blockIdx.x] * mask[threadIdx.x];
}

// ---- fused attention: per (16 i-rows, head, batch) wave --------------------
// logits = (q_h @ k_h^T) + bias + mask  -> LDS -> softmax -> (attn @ v_h)*g
#define LGS 200   // padded row stride (200 % 64 = 8) to break LDS bank conflicts
__global__ void attn_kernel(const float* __restrict__ q, const float* __restrict__ k,
                            const float* __restrict__ v, const float* __restrict__ g,
                            const float* __restrict__ bias, const float* __restrict__ mask,
                            int maskStride, float* __restrict__ hout) {
  __shared__ float lg[16 * LGS];
  const int lane = threadIdx.x;
  const int i0 = blockIdx.x << 4;
  const int h  = blockIdx.y;
  const int b  = blockIdx.z;
  const size_t bs = (size_t)b * NSEQ * HDIM;
  const int am = lane & 15, ah = lane >> 4;
  const float* maskp = mask + (size_t)b * maskStride;

  const float* qrow = q + bs + (size_t)(i0 + am) * HDIM + h * CHEAD;
  const v16bf aq = afrag_row(qrow, 0, ah);             // K = 32 = CHEAD, single step
  for (int jt = 0; jt < NSEQ / 16; ++jt) {
    const int j0 = jt << 4;
    const float* krow = k + bs + (size_t)(j0 + am) * HDIM + h * CHEAD;
    v16bf bk = bfrag_row(krow, 0, ah);
    v8f acc = {};
    WMMA_BF16(acc, aq, bk);
    const int n = am, j = j0 + n;
    const float mk = maskp[j];
#pragma unroll
    for (int r = 0; r < 8; ++r) {
      const int mi = r + (ah << 3);
      const int i = i0 + mi;
      float val = acc[r];
      if (bias) val += bias[((size_t)i * NSEQ + j) * NHEAD + h];
      if (mk < 0.5f) val = NEG_INF;
      lg[mi * LGS + j] = val;
    }
  }
  __syncthreads();
  if (lane < 16) {                                      // softmax over j per row
    float mx = -1e30f;
    for (int j = 0; j < NSEQ; ++j) mx = fmaxf(mx, lg[lane * LGS + j]);
    float sum = 0.f;
    for (int j = 0; j < NSEQ; ++j) { float e = __expf(lg[lane * LGS + j] - mx); lg[lane * LGS + j] = e; sum += e; }
    const float inv = 1.0f / sum;
    for (int j = 0; j < NSEQ; ++j) lg[lane * LGS + j] *= inv;
  }
  __syncthreads();
  // out = attn @ v_h : both 16-column halves off one A fragment per K-step
  v8f acc0 = {}, acc1 = {};
  const float* vcol0 = v + bs + h * CHEAD + am;
  const float* vcol1 = vcol0 + 16;
  for (int k0 = 0; k0 < NSEQ; k0 += 32) {
    v16bf a;
#pragma unroll
    for (int e = 0; e < 8; ++e) {
      a[e]     = (__bf16)lg[am * LGS + k0 + ah * 8 + e];
      a[8 + e] = (__bf16)lg[am * LGS + k0 + 16 + ah * 8 + e];
    }
    v16bf bv0 = bfrag_str(vcol0, k0, ah, HDIM);
    WMMA_BF16(acc0, a, bv0);
    v16bf bv1 = bfrag_str(vcol1, k0, ah, HDIM);
    WMMA_BF16(acc1, a, bv1);
  }
#pragma unroll
  for (int r = 0; r < 8; ++r) {
    const int i = i0 + r + (ah << 3);
    const size_t o = bs + (size_t)i * HDIM + h * CHEAD + am;
    hout[o]      = acc0[r] * g[o];
    hout[o + 16] = acc1[r] * g[o + 16];
  }
}

// ---- outer product pair update: pair += cat(xi*xj, xi-xj) @ wout^T + bout --
// grid = (DPR/64, 12 j-tiles, 192 i); one wave computes 16 j x 64 d
__global__ void outer_kernel(const float* __restrict__ xln, const float* __restrict__ wout,
                             const float* __restrict__ bout, float* __restrict__ outP) {
  __shared__ float xi[DSNG];
  const int lane = threadIdx.x;
  const int n0 = blockIdx.x << 6, j0 = blockIdx.y << 4, i = blockIdx.z;
  for (int t = lane; t < DSNG; t += 32) xi[t] = xln[(size_t)i * DSNG + t];
  __syncthreads();
  const int am = lane & 15, ah = lane >> 4;
  const float* xj = xln + (size_t)(j0 + am) * DSNG;
  const float* wrow0 = wout + (size_t)(n0 + am) * (2 * DSNG);
  v8f z = {};
  v8f acc[4] = {z, z, z, z};
  for (int k0 = 0; k0 < 2 * DSNG; k0 += 32) {
    v16bf a;
#pragma unroll
    for (int e = 0; e < 8; ++e) {
      int k1 = k0 + ah * 8 + e, k2 = k0 + 16 + ah * 8 + e;
      a[e]     = (__bf16)(k1 < DSNG ? xi[k1] * xj[k1] : xi[k1 - DSNG] - xj[k1 - DSNG]);
      a[8 + e] = (__bf16)(k2 < DSNG ? xi[k2] * xj[k2] : xi[k2 - DSNG] - xj[k2 - DSNG]);
    }
#pragma unroll
    for (int t = 0; t < 4; ++t) {
      v16bf b = bfrag_row(wrow0 + (size_t)(t * 16) * (2 * DSNG), k0, ah);
      WMMA_BF16(acc[t], a, b);
    }
  }
#pragma unroll
  for (int t = 0; t < 4; ++t) {
    const int cn = n0 + t * 16 + am;
#pragma unroll
    for (int r = 0; r < 8; ++r) {
      const int j = j0 + r + (ah << 3);
      const size_t o = ((size_t)i * NSEQ + j) * DPR + cn;
      outP[o] += acc[t][r] + bout[cn];
    }
  }
}

// ---- triangle gate*value: abt[d-plane][row] = sig(x@wg+bg)*(x@w+b)*m2[row] -
// stores channel-major planes (256 planes of 192x192) for the per-d matmuls
// one wave: 16 rows x 32 cols of both gate and value (4 WMMAs per K-step)
__global__ void tri_gate_kernel(const float* __restrict__ X,
                                const float* __restrict__ Wg, const float* __restrict__ bg,
                                const float* __restrict__ Wv, const float* __restrict__ bv,
                                const float* __restrict__ m2, float* __restrict__ abt) {
  const int lane = threadIdx.x;
  const int m0 = blockIdx.y << 4, n0 = blockIdx.x << 5;
  const int am = lane & 15, ah = lane >> 4;
  const float* xrow = X + (size_t)(m0 + am) * DPR;
  const float* wgr = Wg + (size_t)(n0 + am) * DPR;
  const float* wvr = Wv + (size_t)(n0 + am) * DPR;
  v8f z = {};
  v8f accg[2] = {z, z}, accv[2] = {z, z};
  for (int k0 = 0; k0 < DPR; k0 += 32) {
    v16bf a = afrag_row(xrow, k0, ah);
#pragma unroll
    for (int t = 0; t < 2; ++t) {
      v16bf b1 = bfrag_row(wgr + (size_t)(t * 16) * DPR, k0, ah);
      WMMA_BF16(accg[t], a, b1);
      v16bf b2 = bfrag_row(wvr + (size_t)(t * 16) * DPR, k0, ah);
      WMMA_BF16(accv[t], a, b2);
    }
  }
#pragma unroll
  for (int t = 0; t < 2; ++t) {
    const int cn = n0 + t * 16 + am;
#pragma unroll
    for (int r = 0; r < 8; ++r) {
      const int m = m0 + r + (ah << 3);
      const float val = sigf(accg[t][r] + bg[cn]) * (accv[t][r] + bv[cn]) * m2[m];
      abt[(size_t)cn * NPAIR + m] = val;
    }
  }
}

// ---- per-channel triangle matmul ------------------------------------------
// transposed==0: t[i,j,d] = sum_k a_d[i,k] * b_d[j,k]   (ikd,jkd->ijd)
// transposed==1: t[i,j,d] = sum_k a_d[k,i] * b_d[k,j]   (kid,kjd->ijd)
// grid = (NSEQ/64, NSEQ/16, DPR); one wave: 16 i x 64 j for one channel d
__global__ void tri_mm_kernel(const float* __restrict__ abt, float* __restrict__ t,
                              int transposed) {
  const int lane = threadIdx.x;
  const int d = blockIdx.z;
  const int m0 = blockIdx.y << 4, n0 = blockIdx.x << 6;
  const int am = lane & 15, ah = lane >> 4;
  const float* pa = abt + (size_t)d * NPAIR;
  const float* pb = abt + (size_t)(DPR + d) * NPAIR;
  v8f z = {};
  v8f acc[4] = {z, z, z, z};
  if (!transposed) {
    const float* arow = pa + (size_t)(m0 + am) * NSEQ;
    const float* brow0 = pb + (size_t)(n0 + am) * NSEQ;
    for (int k0 = 0; k0 < NSEQ; k0 += 32) {
      v16bf a = afrag_row(arow, k0, ah);
#pragma unroll
      for (int tt = 0; tt < 4; ++tt) {
        v16bf b = bfrag_row(brow0 + (size_t)(tt * 16) * NSEQ, k0, ah);
        WMMA_BF16(acc[tt], a, b);
      }
    }
  } else {
    const float* acol = pa + (m0 + am);
    const float* bcol0 = pb + (n0 + am);
    for (int k0 = 0; k0 < NSEQ; k0 += 32) {
      v16bf a = afrag_str(acol, k0, ah, NSEQ);
#pragma unroll
      for (int tt = 0; tt < 4; ++tt) {
        v16bf b = bfrag_str(bcol0 + tt * 16, k0, ah, NSEQ);
        WMMA_BF16(acc[tt], a, b);
      }
    }
  }
#pragma unroll
  for (int tt = 0; tt < 4; ++tt) {
    const int cn = n0 + tt * 16 + am;
#pragma unroll
    for (int r = 0; r < 8; ++r) {
      const int m = m0 + r + (ah << 3);
      t[((size_t)m * NSEQ + cn) * DPR + d] = acc[tt][r];
    }
  }
}

// ---- triangle output: outP += sig(pln@wog+bog) * (tln@wo+bo) ---------------
// one wave: 16 rows x 32 cols of both gate and value (4 WMMAs per K-step)
__global__ void tri_out_kernel(const float* __restrict__ Xg, const float* __restrict__ Wog,
                               const float* __restrict__ bog,
                               const float* __restrict__ Xv, const float* __restrict__ Wo,
                               const float* __restrict__ bo, float* __restrict__ outP) {
  const int lane = threadIdx.x;
  const int m0 = blockIdx.y << 4, n0 = blockIdx.x << 5;
  const int am = lane & 15, ah = lane >> 4;
  const float* xgr = Xg + (size_t)(m0 + am) * DPR;
  const float* xvr = Xv + (size_t)(m0 + am) * DPR;
  const float* wgr = Wog + (size_t)(n0 + am) * DPR;
  const float* wvr = Wo + (size_t)(n0 + am) * DPR;
  v8f z = {};
  v8f accg[2] = {z, z}, accv[2] = {z, z};
  for (int k0 = 0; k0 < DPR; k0 += 32) {
    v16bf ag = afrag_row(xgr, k0, ah);
    v16bf av = afrag_row(xvr, k0, ah);
#pragma unroll
    for (int t = 0; t < 2; ++t) {
      v16bf b1 = bfrag_row(wgr + (size_t)(t * 16) * DPR, k0, ah);
      WMMA_BF16(accg[t], ag, b1);
      v16bf b2 = bfrag_row(wvr + (size_t)(t * 16) * DPR, k0, ah);
      WMMA_BF16(accv[t], av, b2);
    }
  }
#pragma unroll
  for (int t = 0; t < 2; ++t) {
    const int cn = n0 + t * 16 + am;
#pragma unroll
    for (int r = 0; r < 8; ++r) {
      const int m = m0 + r + (ah << 3);
      outP[(size_t)m * DPR + cn] += sigf(accg[t][r] + bog[cn]) * (accv[t][r] + bo[cn]);
    }
  }
}

// ---- pair transpose helpers ------------------------------------------------
__global__ void transposeP_kernel(const float* __restrict__ src, float* __restrict__ dst) {
  const int rc = blockIdx.x;
  const int i = rc / NSEQ, j = rc % NSEQ;
  dst[((size_t)j * NSEQ + i) * DPR + threadIdx.x] = src[(size_t)rc * DPR + threadIdx.x];
}
__global__ void transpose_add_kernel(const float* __restrict__ delta, float* __restrict__ outP) {
  const int rc = blockIdx.x;
  const int i = rc / NSEQ, j = rc % NSEQ;
  outP[(size_t)rc * DPR + threadIdx.x] += delta[((size_t)j * NSEQ + i) * DPR + threadIdx.x];
}

// ---------------------------------------------------------------------------
static void gemm(hipStream_t s, const float* X, int ldx, const float* W, int ldw,
                 const float* bias, float* Y, int ldy, int M, int N, int K,
                 int act, float scale, const float* res) {
  dim3 g(N / 64, M / 16);   // all call sites have N % 64 == 0, M % 16 == 0
  wgemm_kernel<<<g, 32, 0, s>>>(X, ldx, W, ldw, bias, Y, ldy, K, act, scale, res);
}

extern "C" void kernel_launch(void* const* d_in, const int* in_sizes, int n_in,
                              void* d_out, int out_size, void* d_ws, size_t ws_size,
                              hipStream_t stream) {
  (void)in_sizes; (void)n_in; (void)out_size; (void)ws_size;
  const float* single = (const float*)d_in[0];
  const float* pair   = (const float*)d_in[1];
  const float* mask   = (const float*)d_in[2];
  const float* wb     = (const float*)d_in[3];
  const float* bb     = (const float*)d_in[4];
  // sattn
  const float *s_wq = (const float*)d_in[5], *s_wk = (const float*)d_in[6],
              *s_wv = (const float*)d_in[7], *s_wg = (const float*)d_in[8],
              *s_bg = (const float*)d_in[9], *s_wo = (const float*)d_in[10],
              *s_bo = (const float*)d_in[11];
  // sfc
  const float *sf_w1 = (const float*)d_in[12], *sf_b1 = (const float*)d_in[13],
              *sf_w2 = (const float*)d_in[14], *sf_b2 = (const float*)d_in[15];
  const float *wout = (const float*)d_in[16], *bout = (const float*)d_in[17];
  // tm_out: w, b, wg, bg, wo, bo, wog, bog
  const float *to_w = (const float*)d_in[18], *to_b = (const float*)d_in[19],
              *to_wg = (const float*)d_in[20], *to_bg = (const float*)d_in[21],
              *to_wo = (const float*)d_in[22], *to_bo = (const float*)d_in[23],
              *to_wog = (const float*)d_in[24], *to_bog = (const float*)d_in[25];
  const float *ti_w = (const float*)d_in[26], *ti_b = (const float*)d_in[27],
              *ti_wg = (const float*)d_in[28], *ti_bg = (const float*)d_in[29],
              *ti_wo = (const float*)d_in[30], *ti_bo = (const float*)d_in[31],
              *ti_wog = (const float*)d_in[32], *ti_bog = (const float*)d_in[33];
  // pattn_s / pattn_e
  const float *ps_wq = (const float*)d_in[34], *ps_wk = (const float*)d_in[35],
              *ps_wv = (const float*)d_in[36], *ps_wg = (const float*)d_in[37],
              *ps_bg = (const float*)d_in[38], *ps_wo = (const float*)d_in[39],
              *ps_bo = (const float*)d_in[40];
  const float *pe_wq = (const float*)d_in[41], *pe_wk = (const float*)d_in[42],
              *pe_wv = (const float*)d_in[43], *pe_wg = (const float*)d_in[44],
              *pe_bg = (const float*)d_in[45], *pe_wo = (const float*)d_in[46],
              *pe_bo = (const float*)d_in[47];
  // pfc
  const float *pf_w1 = (const float*)d_in[48], *pf_b1 = (const float*)d_in[49],
              *pf_w2 = (const float*)d_in[50], *pf_b2 = (const float*)d_in[51];

  float* outS = (float*)d_out;                  // 192*384
  float* outP = outS + (size_t)NSEQ * DSNG;     // 192*192*128

  // workspace arena (~266 MB of f32); attention and triangle/FC stages alias
  float* ws = (float*)d_ws;
  size_t off = 0;
  auto arena = [&](size_t n) { float* p = ws + off; off += n; return p; };
  float* m2    = arena(NPAIR);
  float* biasb = arena((size_t)NPAIR * NHEAD);
  float* slln  = arena((size_t)NSEQ * DSNG);
  float* pln   = arena((size_t)NPAIR * DPR);
  float* qb    = arena((size_t)NPAIR * HDIM);
  float* kb    = arena((size_t)NPAIR * HDIM);
  float* vb    = arena((size_t)NPAIR * HDIM);
  float* gb    = arena((size_t)NPAIR * HDIM);
  float* houtb = arena((size_t)NPAIR * HDIM);
  float* tbuf  = arena((size_t)NPAIR * DPR);
  float* tln   = arena((size_t)NPAIR * DPR);
  float* pairT = arena((size_t)NPAIR * DPR);
  float* abt    = qb;     // 256 planes * 36864 = NPAIR*HDIM, tri stages only
  float* fc1    = qb;     // up to NPAIR*512 spans qb+kb, FC stages only
  float* deltaT = tbuf;   // pattn_e delta, tri stages finished by then

  const float qscale = 0.17677669529663687f;    // 1/sqrt(C)

  // residual accumulators start as inputs
  hipMemcpyAsync(outS, single, sizeof(float) * (size_t)NSEQ * DSNG, hipMemcpyDeviceToDevice, stream);
  hipMemcpyAsync(outP, pair, sizeof(float) * (size_t)NPAIR * DPR, hipMemcpyDeviceToDevice, stream);
  mask2d_kernel<<<dim3(NSEQ), dim3(NSEQ), 0, stream>>>(mask, m2);

  // attention bias from ORIGINAL pair: bias[(i*N+j)*H + h]
  ln_kernel<<<dim3(NPAIR), dim3(128), 0, stream>>>(pair, pln, DPR);
  bias_proj_kernel<<<dim3((NPAIR * NHEAD + 255) / 256), dim3(256), 0, stream>>>(pln, wb, bb, biasb);

  // ---- single attention ----
  ln_kernel<<<dim3(NSEQ), dim3(128), 0, stream>>>(outS, slln, DSNG);
  gemm(stream, slln, DSNG, s_wq, DSNG, nullptr, qb, HDIM, NSEQ, HDIM, DSNG, 0, qscale, nullptr);
  gemm(stream, slln, DSNG, s_wk, DSNG, nullptr, kb, HDIM, NSEQ, HDIM, DSNG, 0, 1.f, nullptr);
  gemm(stream, slln, DSNG, s_wv, DSNG, nullptr, vb, HDIM, NSEQ, HDIM, DSNG, 0, 1.f, nullptr);
  gemm(stream, slln, DSNG, s_wg, DSNG, s_bg, gb, HDIM, NSEQ, HDIM, DSNG, 2, 1.f, nullptr);
  attn_kernel<<<dim3(NSEQ / 16, NHEAD, 1), 32, 0, stream>>>(qb, kb, vb, gb, biasb, mask, 0, houtb);
  gemm(stream, houtb, HDIM, s_wo, HDIM, s_bo, outS, DSNG, NSEQ, DSNG, HDIM, 0, 1.f, outS);

  // ---- single transition ----
  ln_kernel<<<dim3(NSEQ), dim3(128), 0, stream>>>(outS, slln, DSNG);
  gemm(stream, slln, DSNG, sf_w1, DSNG, sf_b1, fc1, 4 * DSNG, NSEQ, 4 * DSNG, DSNG, 1, 1.f, nullptr);
  gemm(stream, fc1, 4 * DSNG, sf_w2, 4 * DSNG, sf_b2, outS, DSNG, NSEQ, DSNG, 4 * DSNG, 0, 1.f, outS);

  // ---- outer product pair update ----
  ln_kernel<<<dim3(NSEQ), dim3(128), 0, stream>>>(outS, slln, DSNG);
  outer_kernel<<<dim3(DPR / 64, NSEQ / 16, NSEQ), 32, 0, stream>>>(slln, wout, bout, outP);

  // ---- triangle multiplication (outgoing: ikd,jkd->ijd) ----
  ln_kernel<<<dim3(NPAIR), dim3(128), 0, stream>>>(outP, pln, DPR);
  tri_gate_kernel<<<dim3(8, NPAIR / 16), 32, 0, stream>>>(pln, to_wg, to_bg, to_w, to_b, m2, abt);
  tri_mm_kernel<<<dim3(NSEQ / 64, NSEQ / 16, DPR), 32, 0, stream>>>(abt, tbuf, 0);
  ln_kernel<<<dim3(NPAIR), dim3(128), 0, stream>>>(tbuf, tln, DPR);
  tri_out_kernel<<<dim3(DPR / 32, NPAIR / 16), 32, 0, stream>>>(pln, to_wog, to_bog, tln, to_wo, to_bo, outP);

  // ---- triangle multiplication (incoming: kid,kjd->ijd) ----
  ln_kernel<<<dim3(NPAIR), dim3(128), 0, stream>>>(outP, pln, DPR);
  tri_gate_kernel<<<dim3(8, NPAIR / 16), 32, 0, stream>>>(pln, ti_wg, ti_bg, ti_w, ti_b, m2, abt);
  tri_mm_kernel<<<dim3(NSEQ / 64, NSEQ / 16, DPR), 32, 0, stream>>>(abt, tbuf, 1);
  ln_kernel<<<dim3(NPAIR), dim3(128), 0, stream>>>(tbuf, tln, DPR);
  tri_out_kernel<<<dim3(DPR / 32, NPAIR / 16), 32, 0, stream>>>(pln, ti_wog, ti_bog, tln, ti_wo, ti_bo, outP);

  // ---- pair attention, row-wise (starting node) ----
  ln_kernel<<<dim3(NPAIR), dim3(128), 0, stream>>>(outP, pln, DPR);
  gemm(stream, pln, DPR, ps_wq, DPR, nullptr, qb, HDIM, NPAIR, HDIM, DPR, 0, qscale, nullptr);
  gemm(stream, pln, DPR, ps_wk, DPR, nullptr, kb, HDIM, NPAIR, HDIM, DPR, 0, 1.f, nullptr);
  gemm(stream, pln, DPR, ps_wv, DPR, nullptr, vb, HDIM, NPAIR, HDIM, DPR, 0, 1.f, nullptr);
  gemm(stream, pln, DPR, ps_wg, DPR, ps_bg, gb, HDIM, NPAIR, HDIM, DPR, 2, 1.f, nullptr);
  attn_kernel<<<dim3(NSEQ / 16, NHEAD, NSEQ), 32, 0, stream>>>(qb, kb, vb, gb, nullptr, m2, NSEQ, houtb);
  gemm(stream, houtb, HDIM, ps_wo, HDIM, ps_bo, outP, DPR, NPAIR, DPR, HDIM, 0, 1.f, outP);

  // ---- pair attention, column-wise (ending node) on transposed pair ----
  transposeP_kernel<<<dim3(NPAIR), dim3(DPR), 0, stream>>>(outP, pairT);
  ln_kernel<<<dim3(NPAIR), dim3(128), 0, stream>>>(pairT, pln, DPR);
  gemm(stream, pln, DPR, pe_wq, DPR, nullptr, qb, HDIM, NPAIR, HDIM, DPR, 0, qscale, nullptr);
  gemm(stream, pln, DPR, pe_wk, DPR, nullptr, kb, HDIM, NPAIR, HDIM, DPR, 0, 1.f, nullptr);
  gemm(stream, pln, DPR, pe_wv, DPR, nullptr, vb, HDIM, NPAIR, HDIM, DPR, 0, 1.f, nullptr);
  gemm(stream, pln, DPR, pe_wg, DPR, pe_bg, gb, HDIM, NPAIR, HDIM, DPR, 2, 1.f, nullptr);
  // mask2d is symmetric (outer product of mask), so the transposed mask == m2
  attn_kernel<<<dim3(NSEQ / 16, NHEAD, NSEQ), 32, 0, stream>>>(qb, kb, vb, gb, nullptr, m2, NSEQ, houtb);
  gemm(stream, houtb, HDIM, pe_wo, HDIM, pe_bo, deltaT, DPR, NPAIR, DPR, HDIM, 0, 1.f, nullptr);
  transpose_add_kernel<<<dim3(NPAIR), dim3(DPR), 0, stream>>>(deltaT, outP);

  // ---- pair transition ----
  ln_kernel<<<dim3(NPAIR), dim3(128), 0, stream>>>(outP, pln, DPR);
  gemm(stream, pln, DPR, pf_w1, DPR, pf_b1, fc1, 4 * DPR, NPAIR, 4 * DPR, DPR, 1, 1.f, nullptr);
  gemm(stream, fc1, 4 * DPR, pf_w2, 4 * DPR, pf_b2, outP, DPR, NPAIR, DPR, 4 * DPR, 0, 1.f, outP);
}
